// DecomposingAttnProcessor_pad_41669772706191
// MI455X (gfx1250) — matrix operand order repaired and checked by reading, not compiled, not run on hardware
//
#include <hip/hip_runtime.h>

// ---------------------------------------------------------------------------
// Types for CDNA5 WMMA (gfx1250, wave32)
// ---------------------------------------------------------------------------
typedef __bf16 bf16;
typedef __attribute__((ext_vector_type(16))) __bf16 v16bf;
typedef __attribute__((ext_vector_type(8)))  float  v8f;

union V16U { uint4 u[2]; v16bf v; };

// WMMA A-operand slice: 8 contiguous bf16 from p0 (elems 0..7) and 8 from p1
// (elems 8..15). Two b128 loads.
__device__ __forceinline__ v16bf ld_split16(const bf16* p0, const bf16* p1) {
    V16U x;
    x.u[0] = *(const uint4*)p0;
    x.u[1] = *(const uint4*)p1;
    return x.v;
}

// 16 contiguous bf16 (32 bytes) as two b128 loads.
__device__ __forceinline__ v16bf ld_contig16(const bf16* p) {
    V16U x;
    const uint4* q = (const uint4*)p;
    x.u[0] = q[0];
    x.u[1] = q[1];
    return x.v;
}

__device__ __forceinline__ v8f wmma_bf16(v16bf a, v16bf b, v8f c) {
    // D = A(16x32 bf16) * B(32x16 bf16) + C(16x16 f32)
    return __builtin_amdgcn_wmma_f32_16x16x32_bf16(
        /*neg_a=*/false, a, /*neg_b=*/false, b,
        /*c_mod=*/(short)0, c, /*reuse_a=*/false, /*reuse_b=*/false);
}

__device__ __forceinline__ v8f v8f_zero() {
    return (v8f){0.f, 0.f, 0.f, 0.f, 0.f, 0.f, 0.f, 0.f};
}

// ---------------------------------------------------------------------------
// Problem constants (fixed by the reference harness)
// ---------------------------------------------------------------------------
#define NC   4
#define BB   2        // batch (BTC / NC)
#define BTC  8
#define SQ   2048     // query tokens
#define DD   1024     // hidden dim
#define EE   256      // encoder tokens
#define HH   16       // heads
#define DH   64       // head dim

// ---------------------------------------------------------------------------
// fp32 -> bf16 conversion (grid-stride)
// ---------------------------------------------------------------------------
__global__ void f2bf_kernel(const float* __restrict__ x, bf16* __restrict__ y,
                            long long n) {
    long long i = (long long)blockIdx.x * blockDim.x + threadIdx.x;
    long long stride = (long long)gridDim.x * blockDim.x;
    for (; i < n; i += stride) y[i] = (bf16)x[i];
}

// Weight transpose + convert: W[k][n] (1024x1024 f32) -> Wt[n][k] bf16
__global__ void wtr_kernel(const float* __restrict__ w, bf16* __restrict__ wt) {
    int i = blockIdx.x * blockDim.x + threadIdx.x;   // one elem per thread
    int k = i >> 10;
    int n = i & 1023;
    wt[(size_t)n * 1024 + k] = (bf16)w[i];
}

// ---------------------------------------------------------------------------
// Generic WMMA GEMM: C[M,N] = A[M,K] @ Wt[N,K]^T  (bf16 in, f32 accum)
// Wave computes 32(M) x 64(N) (8 WMMAs / k-step); WG = 4 waves stacked in M.
// k-loop unrolled x2 with ping-pong operand buffers: no register rotation
// copies, loads for step i+1 issue while step i's WMMAs execute.
// mode 0: store bf16 row-major C[M,N]
// mode 1: store bf16 "V transposed": m = btc*256+e, out[btc][n][e] (Vt layout)
// mode 2: store f32  C[M,N] + resid[M,N]   (final projection + residual)
// ---------------------------------------------------------------------------
__global__ __launch_bounds__(128)
void gemm_bf16_kernel(const bf16* __restrict__ A, const bf16* __restrict__ Wt,
                      bf16* __restrict__ outB, float* __restrict__ outF,
                      const float* __restrict__ resid,
                      int M, int N, int K, int mode) {
    const int lane = threadIdx.x & 31;
    const int wave = threadIdx.x >> 5;
    const int lm   = lane & 15;        // row/col index within 16
    const int hi   = lane >> 4;        // 0 or 1 (lane half)

    const int m0 = (blockIdx.y * 4 + wave) * 32;   // 32 M-rows per wave
    const int n0 = blockIdx.x * 64;
    if (m0 >= M) return;

    v8f acc[2][4];
#pragma unroll
    for (int mi = 0; mi < 2; ++mi)
#pragma unroll
        for (int j = 0; j < 4; ++j) acc[mi][j] = v8f_zero();

    const bf16* ar0 = A + (size_t)(m0 + lm) * K + hi * 16;
    const bf16* ar1 = A + (size_t)(m0 + 16 + lm) * K + hi * 16;
    const bf16* br0 = Wt + (size_t)(n0 + 0 * 16 + lm) * K + hi * 16;
    const bf16* br1 = Wt + (size_t)(n0 + 1 * 16 + lm) * K + hi * 16;
    const bf16* br2 = Wt + (size_t)(n0 + 2 * 16 + lm) * K + hi * 16;
    const bf16* br3 = Wt + (size_t)(n0 + 3 * 16 + lm) * K + hi * 16;

#define LOAD_STEP(aa, bb, k)                                                   \
    do {                                                                       \
        aa[0] = ld_split16(ar0 + (k), ar0 + (k) + 8);                          \
        aa[1] = ld_split16(ar1 + (k), ar1 + (k) + 8);                          \
        bb[0] = ld_contig16(br0 + (k));                                        \
        bb[1] = ld_contig16(br1 + (k));                                        \
        bb[2] = ld_contig16(br2 + (k));                                        \
        bb[3] = ld_contig16(br3 + (k));                                        \
    } while (0)

#define WMMA_STEP(aa, bb)                                                      \
    do {                                                                       \
        _Pragma("unroll") for (int j = 0; j < 4; ++j) {                        \
            acc[0][j] = wmma_bf16(aa[0], bb[j], acc[0][j]);                    \
            acc[1][j] = wmma_bf16(aa[1], bb[j], acc[1][j]);                    \
        }                                                                      \
    } while (0)

    v16bf a0[2], b0[4], a1[2], b1[4];
    LOAD_STEP(a0, b0, 0);                    // k-step 0

    // Ping-pong main loop: covers k pairs (k0, k0+32) for k0 = 0..K-128.
    for (int k0 = 0; k0 < K - 64; k0 += 64) {
        LOAD_STEP(a1, b1, k0 + 32);          // prefetch odd step
        WMMA_STEP(a0, b0);                   // compute even step
        LOAD_STEP(a0, b0, k0 + 64);          // prefetch next even step
        WMMA_STEP(a1, b1);                   // compute odd step
    }
    // Epilogue pair: k = K-64, K-32.
    LOAD_STEP(a1, b1, K - 32);
    WMMA_STEP(a0, b0);
    WMMA_STEP(a1, b1);

#undef LOAD_STEP
#undef WMMA_STEP

    // C layout: VGPR r -> row (r + 8*hi), col = lm
#pragma unroll
    for (int mi = 0; mi < 2; ++mi) {
#pragma unroll
        for (int j = 0; j < 4; ++j) {
#pragma unroll
            for (int r = 0; r < 8; ++r) {
                const int m = m0 + mi * 16 + r + hi * 8;
                const int n = n0 + j * 16 + lm;
                const float v = acc[mi][j][r];
                if (mode == 0) {
                    outB[(size_t)m * N + n] = (bf16)v;
                } else if (mode == 1) {
                    const int btc = m >> 8;           // m = btc*256 + e
                    const int e   = m & 255;
                    outB[((size_t)btc * N + n) * EE + e] = (bf16)v;
                } else {
                    const size_t idx = (size_t)m * N + n;
                    outF[idx] = v + resid[idx];
                }
            }
        }
    }
}

// ---------------------------------------------------------------------------
// Fused decomposing attention.
//   grid = (S/64, H, B), block = 128 (4 waves). Wave owns one 16-row s-tile
//   of one (b,h) for ALL 4 components (the softmax couples only components,
//   never E -> E is freely tiled in 32-wide chunks).
//   Per 32-wide e-chunk:
//     scores_c = Q_c K_c^T        (loads batched before each WMMA group)
//     elementwise component-softmax + entropy (written straight to d_out)
//     all w_c -> bf16 -> LDS stage (C->A relayout), then per component:
//     2x ds_load_b128 + 4x V-loads, then 4 WMMAs (out_c += w_c @ V_c)
// ---------------------------------------------------------------------------
__global__ __launch_bounds__(128)
void attn_kernel(const bf16* __restrict__ Qb, const bf16* __restrict__ Kb,
                 const bf16* __restrict__ Vt, bf16* __restrict__ attnOut,
                 float* __restrict__ entOut) {
    __shared__ bf16 stage[4][NC][16 * 32];   // 1KB per wave per component

    const int lane = threadIdx.x & 31;
    const int wave = threadIdx.x >> 5;
    const int lm   = lane & 15;
    const int hi   = lane >> 4;

    const int s0 = blockIdx.x * 64 + wave * 16;
    const int h  = blockIdx.y;
    const int b  = blockIdx.z;
    const float scale = 0.125f;          // 1/sqrt(64)

    size_t qbase[NC], kbase[NC], vbase[NC];
#pragma unroll
    for (int c = 0; c < NC; ++c) {
        const int btc = c * BB + b;
        qbase[c] = ((size_t)btc * SQ + s0 + lm) * DD + h * DH + hi * 16;
        kbase[c] = (size_t)btc * EE * DD + h * DH + hi * 16;
        vbase[c] = ((size_t)btc * DD + h * DH) * EE + hi * 16;
    }

    v8f o[NC][4];
#pragma unroll
    for (int c = 0; c < NC; ++c)
#pragma unroll
        for (int j = 0; j < 4; ++j) o[c][j] = v8f_zero();

    for (int ec = 0; ec < EE / 32; ++ec) {
        const int e0 = ec * 32;

        // ---- scores for all 4 components, two 16x16 e-tiles each ----
        v8f sc[NC][2];
#pragma unroll
        for (int c = 0; c < NC; ++c) {
            const bf16* qr = Qb + qbase[c];
            const bf16* kp0 = Kb + kbase[c] + (size_t)(e0 + lm) * DD;
            const bf16* kp1 = kp0 + (size_t)16 * DD;
            // batch all loads for this component before its WMMA group
            v16bf a0 = ld_split16(qr, qr + 8);             // k 0..31
            v16bf a1 = ld_split16(qr + 32, qr + 40);       // k 32..63
            v16bf b00 = ld_contig16(kp0);                  // t=0, k 0..31
            v16bf b01 = ld_contig16(kp0 + 32);             // t=0, k 32..63
            v16bf b10 = ld_contig16(kp1);                  // t=1, k 0..31
            v16bf b11 = ld_contig16(kp1 + 32);             // t=1, k 32..63
            v8f s0v = v8f_zero(), s1v = v8f_zero();
            s0v = wmma_bf16(a0, b00, s0v);
            s0v = wmma_bf16(a1, b01, s0v);
            s1v = wmma_bf16(a0, b10, s1v);
            s1v = wmma_bf16(a1, b11, s1v);
            sc[c][0] = s0v;
            sc[c][1] = s1v;
        }

        // ---- component softmax + entropy (elementwise on C layout) ----
#pragma unroll
        for (int t = 0; t < 2; ++t) {
#pragma unroll
            for (int r = 0; r < 8; ++r) {
                float l0 = sc[0][t][r] * scale;
                float l1 = sc[1][t][r] * scale;
                float l2 = sc[2][t][r] * scale;
                float l3 = sc[3][t][r] * scale;
                float mx = fmaxf(fmaxf(l0, l1), fmaxf(l2, l3));
                float p0 = __expf(l0 - mx), p1 = __expf(l1 - mx);
                float p2 = __expf(l2 - mx), p3 = __expf(l3 - mx);
                float Z  = p0 + p1 + p2 + p3;
                float iv = 1.0f / Z;
                float w0 = p0 * iv, w1 = p1 * iv, w2 = p2 * iv, w3 = p3 * iv;
                float lnZ = __logf(Z);
                float ent = w0 * ((l0 - mx) - lnZ) + w1 * ((l1 - mx) - lnZ)
                          + w2 * ((l2 - mx) - lnZ) + w3 * ((l3 - mx) - lnZ);
                const int mr = r + hi * 8;
                entOut[((size_t)(b * HH + h) * SQ + (s0 + mr)) * EE
                       + e0 + t * 16 + lm] = ent;
                sc[0][t][r] = w0; sc[1][t][r] = w1;
                sc[2][t][r] = w2; sc[3][t][r] = w3;
            }
        }

        // ---- stage ALL components' w tiles to LDS (C-layout -> row-major) --
#pragma unroll
        for (int c = 0; c < NC; ++c)
#pragma unroll
            for (int t = 0; t < 2; ++t)
#pragma unroll
                for (int r = 0; r < 8; ++r)
                    stage[wave][c][(r + hi * 8) * 32 + t * 16 + lm] =
                        (bf16)sc[c][t][r];

        // ---- out_c += w_c @ V_c (ds + global loads batched per component) --
#pragma unroll
        for (int c = 0; c < NC; ++c) {
            const bf16* sp = &stage[wave][c][lm * 32 + hi * 16];
            v16bf aw = ld_split16(sp, sp + 8);        // A operand, K = 32 (e)
            v16bf vb[4];
#pragma unroll
            for (int j = 0; j < 4; ++j)
                vb[j] = ld_contig16(Vt + vbase[c] + (size_t)(j * 16 + lm) * EE
                                    + e0);
#pragma unroll
            for (int j = 0; j < 4; ++j)
                o[c][j] = wmma_bf16(aw, vb[j], o[c][j]);
        }
    }

    // ---- store attention output (bf16, [BTC,S,D] row-major) ----
#pragma unroll
    for (int c = 0; c < NC; ++c) {
        const int btc = c * BB + b;
#pragma unroll
        for (int j = 0; j < 4; ++j)
#pragma unroll
            for (int r = 0; r < 8; ++r)
                attnOut[((size_t)btc * SQ + s0 + r + hi * 8) * DD
                        + h * DH + j * 16 + lm] = (bf16)o[c][j][r];
    }
}

// ---------------------------------------------------------------------------
// Launcher
// ---------------------------------------------------------------------------
extern "C" void kernel_launch(void* const* d_in, const int* in_sizes, int n_in,
                              void* d_out, int out_size, void* d_ws,
                              size_t ws_size, hipStream_t stream) {
    const float* hid = (const float*)d_in[0];   // [8,2048,1024]
    const float* enc = (const float*)d_in[1];   // [8,256,1024]
    const float* Wq  = (const float*)d_in[2];
    const float* Wk  = (const float*)d_in[3];
    const float* Wv  = (const float*)d_in[4];
    const float* Wo  = (const float*)d_in[5];
    // d_in[6] = pad_length (dead: pad stream contributes zero to both outputs)

    float* out = (float*)d_out;                       // [8,2048,1024]
    float* ent = out + (size_t)BTC * SQ * DD;         // [2,16,2048,256]

    const long long N_HID = (long long)BTC * SQ * DD; // 16,777,216
    const long long N_ENC = (long long)BTC * EE * DD; //  2,097,152

    // Workspace carve-up (bf16 buffers), ~84 MB total.
    char* ws = (char*)d_ws;
    size_t off = 0;
    bf16* hid_bf = (bf16*)(ws + off); off += (size_t)N_HID * 2;  // 33.5MB
    bf16* enc_bf = (bf16*)(ws + off); off += (size_t)N_ENC * 2;  //  4.2MB
    bf16* wq_t   = (bf16*)(ws + off); off += (size_t)DD * DD * 2;
    bf16* wk_t   = (bf16*)(ws + off); off += (size_t)DD * DD * 2;
    bf16* wv_t   = (bf16*)(ws + off); off += (size_t)DD * DD * 2;
    bf16* wo_t   = (bf16*)(ws + off); off += (size_t)DD * DD * 2;
    bf16* q_bf   = (bf16*)(ws + off); off += (size_t)N_HID * 2;  // 33.5MB
    bf16* k_bf   = (bf16*)(ws + off); off += (size_t)N_ENC * 2;
    bf16* vt_bf  = (bf16*)(ws + off); off += (size_t)N_ENC * 2;
    bf16* attn_bf = hid_bf;  // hid_bf is dead after the Q projection; reuse.

    // 1) convert activations, convert+transpose weights
    f2bf_kernel<<<4096, 256, 0, stream>>>(hid, hid_bf, N_HID);
    f2bf_kernel<<<2048, 256, 0, stream>>>(enc, enc_bf, N_ENC);
    wtr_kernel<<<4096, 256, 0, stream>>>(Wq, wq_t);
    wtr_kernel<<<4096, 256, 0, stream>>>(Wk, wk_t);
    wtr_kernel<<<4096, 256, 0, stream>>>(Wv, wv_t);
    wtr_kernel<<<4096, 256, 0, stream>>>(Wo, wo_t);

    // 2) projections (wave tile 32x64, WG tile 128x64)
    gemm_bf16_kernel<<<dim3(16, 128), 128, 0, stream>>>(
        hid_bf, wq_t, q_bf, nullptr, nullptr, BTC * SQ, DD, DD, 0);
    gemm_bf16_kernel<<<dim3(16, 16), 128, 0, stream>>>(
        enc_bf, wk_t, k_bf, nullptr, nullptr, BTC * EE, DD, DD, 0);
    gemm_bf16_kernel<<<dim3(16, 16), 128, 0, stream>>>(
        enc_bf, wv_t, vt_bf, nullptr, nullptr, BTC * EE, DD, DD, 1);

    // 3) fused component-softmax attention (+ entropy output)
    attn_kernel<<<dim3(SQ / 64, HH, BB), 128, 0, stream>>>(
        q_bf, k_bf, vt_bf, attn_bf, ent);

    // 4) output projection + residual -> d_out (fp32)
    gemm_bf16_kernel<<<dim3(16, 128), 128, 0, stream>>>(
        attn_bf, wo_t, nullptr, out, hid, BTC * SQ, DD, DD, 2);
}
